// RowAttentionWithPairBias_36172214566956
// MI455X (gfx1250) — matrix operand order repaired and checked by reading, not compiled, not run on hardware
//
#include <hip/hip_runtime.h>

typedef __attribute__((ext_vector_type(16))) _Float16 v16h;
typedef __attribute__((ext_vector_type(8)))  float    v8f;

// ---------------------------------------------------------------------------
// WMMA fragment helpers (CDNA5 gfx1250, wave32, 16x16x32 f16 -> f32)
//
// A-matrix 16x32 f16 layout (ISA 7.12.2): lane L: m = L%16, half = L/16.
//   VGPR j<4 : K = half*8 + 2j, 2j+1      VGPR j>=4: K = 16 + half*8 + 2(j-4)
// B-matrix 32x16 f16: stored [N][K] row-major; lane L: n = L%16, half = L/16.
//   VGPR j<4 : K = half*8 + 2j, 2j+1      VGPR j>=4: K = 16 + half*8 + 2(j-4)
// C/D f32 16x16: lane L: n = L%16; VGPR r: m = r + 8*(L/16)
// ---------------------------------------------------------------------------
__device__ __forceinline__ v16h load_frag_a(const _Float16* base, int ld) {
  const int lane = threadIdx.x & 31;
  const int m = lane & 15, hf = lane >> 4;
  union { v16h v; unsigned int u[8]; } r;
  const char* b = (const char*)(base + m * ld);
#pragma unroll
  for (int j = 0; j < 8; ++j) {
    int k = (j < 4) ? (hf * 8 + 2 * j) : (16 + hf * 8 + 2 * (j - 4));
    r.u[j] = *(const unsigned int*)(b + 2 * k);
  }
  return r.v;
}

__device__ __forceinline__ v16h load_frag_b(const _Float16* base, int ld) {
  const int lane = threadIdx.x & 31;
  const int n = lane & 15, hf = lane >> 4;
  union { v16h v; unsigned int u[8]; } r;
  const char* b = (const char*)(base + n * ld);
#pragma unroll
  for (int j = 0; j < 8; ++j) {
    int k = (j < 4) ? (hf * 8 + 2 * j) : (16 + hf * 8 + 2 * (j - 4));
    r.u[j] = *(const unsigned int*)(b + 2 * k);
  }
  return r.v;
}

__device__ __forceinline__ v8f wmma_f16(v16h a, v16h b, v8f c) {
  return __builtin_amdgcn_wmma_f32_16x16x32_f16(false, a, false, b, (short)0, c,
                                                false, false);
}

// ---------------------------------------------------------------------------
// Async global -> LDS 16-byte copy (CDNA5 GLOBAL_LOAD_ASYNC_TO_LDS_B128,
// GVS addressing: 64-bit SGPR base + 32-bit per-lane offset; VDST is the
// per-lane LDS byte address).  Tracked on ASYNCcnt.
// ---------------------------------------------------------------------------
__device__ __forceinline__ void async_copy16(unsigned lds_addr,
                                             unsigned long long gbase,
                                             unsigned goff) {
  asm volatile("global_load_async_to_lds_b128 %0, %1, %2"
               :
               : "v"(lds_addr), "v"(goff), "s"(gbase)
               : "memory");
}

__device__ __forceinline__ void wait_async0() {
  asm volatile("s_wait_asynccnt 0" ::: "memory");
}

__device__ __forceinline__ unsigned lds_byte_addr(const void* p) {
  // Generic LDS pointer: low 32 bits are the LDS byte address (ISA 10.2).
  return (unsigned)(unsigned long long)(size_t)p;
}

// ---------------------------------------------------------------------------
// Kernel 1: weight conversion.  wq/wk/wv/wg (C_M=256, H=8, C=32) f32 ->
// f16 [h][d][c]  (B-fragment order [N][K]).  wo (8,32,256) -> f16 [c][h*32+d].
// ---------------------------------------------------------------------------
__global__ __launch_bounds__(256) void prep_weights_kernel(
    const float* __restrict__ wq, const float* __restrict__ wk,
    const float* __restrict__ wv, const float* __restrict__ wg,
    const float* __restrict__ wo,
    _Float16* __restrict__ wq_t, _Float16* __restrict__ wk_t,
    _Float16* __restrict__ wv_t, _Float16* __restrict__ wg_t,
    _Float16* __restrict__ wo_t) {
  int i = blockIdx.x * 256 + threadIdx.x;   // 5 * 65536 threads
  int seg = i >> 16, j = i & 65535;
  if (seg < 4) {
    const float* src = (seg == 0) ? wq : (seg == 1) ? wk : (seg == 2) ? wv : wg;
    _Float16* dst = (seg == 0) ? wq_t : (seg == 1) ? wk_t : (seg == 2) ? wv_t : wg_t;
    int h = j >> 13;             // j = h*8192 + d*256 + c
    int d = (j >> 8) & 31;
    int c = j & 255;
    dst[j] = (_Float16)src[c * 256 + h * 32 + d];
  } else {
    int c = j >> 8, k = j & 255; // k = h*32 + d
    wo_t[j] = (_Float16)wo[k * 256 + c];
  }
}

// ---------------------------------------------------------------------------
// Kernel 2: pair bias.  One wave per (q,k): layernorm over C_Z=128, then
// project to H=8 heads.  z[h][q][k] f32.
// ---------------------------------------------------------------------------
__global__ __launch_bounds__(256) void pair_bias_kernel(
    const float* __restrict__ pair_act, const float* __restrict__ pn_g,
    const float* __restrict__ pn_b, const float* __restrict__ w2d,
    float* __restrict__ z_buf) {
  const int w = threadIdx.x >> 5, lane = threadIdx.x & 31;
  const int p = blockIdx.x * 8 + w;          // 0..65535  (q*256 + k)
  const float* x = pair_act + (size_t)p * 128;
  float v[4], sum = 0.f, sq = 0.f;
#pragma unroll
  for (int j = 0; j < 4; ++j) {
    v[j] = x[lane + 32 * j];
    sum += v[j];
    sq += v[j] * v[j];
  }
#pragma unroll
  for (int off = 16; off; off >>= 1) {
    sum += __shfl_xor(sum, off, 32);
    sq += __shfl_xor(sq, off, 32);
  }
  float mu = sum * (1.f / 128.f);
  float var = sq * (1.f / 128.f) - mu * mu;
  float rs = rsqrtf(var + 1e-5f);
  float xn[4];
#pragma unroll
  for (int j = 0; j < 4; ++j) {
    int c = lane + 32 * j;
    xn[j] = (v[j] - mu) * rs * pn_g[c] + pn_b[c];
  }
  const int q = p >> 8, kk = p & 255;
#pragma unroll
  for (int hh = 0; hh < 8; ++hh) {
    float acc = 0.f;
#pragma unroll
    for (int j = 0; j < 4; ++j) acc += xn[j] * w2d[(lane + 32 * j) * 8 + hh];
#pragma unroll
    for (int off = 16; off; off >>= 1) acc += __shfl_xor(acc, off, 32);
    if (lane == 0) z_buf[(hh * 256 + q) * 256 + kk] = acc;
  }
}

// ---------------------------------------------------------------------------
// Kernel 3: layernorm of msa_act -> m f16 [row=(s*256+q)][c]
// ---------------------------------------------------------------------------
__global__ __launch_bounds__(256) void ln_msa_kernel(
    const float* __restrict__ msa_act, const float* __restrict__ g,
    const float* __restrict__ b, _Float16* __restrict__ m_buf) {
  const int w = threadIdx.x >> 5, lane = threadIdx.x & 31;
  const int row = blockIdx.x * 8 + w;        // 0..32767
  const float* x = msa_act + (size_t)row * 256;
  float v[8], sum = 0.f, sq = 0.f;
#pragma unroll
  for (int j = 0; j < 8; ++j) {
    v[j] = x[lane + 32 * j];
    sum += v[j];
    sq += v[j] * v[j];
  }
#pragma unroll
  for (int off = 16; off; off >>= 1) {
    sum += __shfl_xor(sum, off, 32);
    sq += __shfl_xor(sq, off, 32);
  }
  float mu = sum * (1.f / 256.f);
  float var = sq * (1.f / 256.f) - mu * mu;
  float rs = rsqrtf(var + 1e-5f);
#pragma unroll
  for (int j = 0; j < 8; ++j) {
    int c = lane + 32 * j;
    m_buf[(size_t)row * 256 + c] = (_Float16)((v[j] - mu) * rs * g[c] + b[c]);
  }
}

// ---------------------------------------------------------------------------
// Kernel 4: fused attention.  grid = S*H = 1024 WGs, 256 threads (8 wave32).
// LDS (dynamic, 228864 B):
//   mp   : f16 256x264  (m[s] staged; later reused as probs)   @ 0
//   qs   : f16 256x40                                           @ 135168
//   ks   : f16 256x40                                           @ 155648
//   vt   : f16  32x264  (V transposed: [d][key])                @ 176128
//   gate : f32 256x33                                           @ 193024
//   rowmax f32 256                                              @ 226816
//   rowsum f32 256                                              @ 227840
// ---------------------------------------------------------------------------
#define ATTN_LDS_BYTES 228864

__global__ __launch_bounds__(256) void attn_kernel(
    const float* __restrict__ msa_mask, const float* __restrict__ bg,
    const _Float16* __restrict__ m_buf, const float* __restrict__ z_buf,
    const _Float16* __restrict__ wq_t, const _Float16* __restrict__ wk_t,
    const _Float16* __restrict__ wv_t, const _Float16* __restrict__ wg_t,
    _Float16* __restrict__ o_buf) {
  extern __shared__ uint4 dyn_smem[];
  char* smem = (char*)dyn_smem;
  _Float16* mp = (_Float16*)(smem);
  _Float16* qs = (_Float16*)(smem + 135168);
  _Float16* ks = (_Float16*)(smem + 155648);
  _Float16* vt = (_Float16*)(smem + 176128);
  float* gate = (float*)(smem + 193024);
  float* rowmax = (float*)(smem + 226816);
  float* rowsum = (float*)(smem + 227840);

  const int s = blockIdx.x >> 3;
  const int h = blockIdx.x & 7;
  const int tid = threadIdx.x;
  const int w = tid >> 5, lane = tid & 31;
  const int lm = lane & 15, hf = lane >> 4;

  // ---- phase 0: async-stage m[s] (256x256 f16) into mp, padded ld=264 ----
  {
    const unsigned long long gbase =
        (unsigned long long)(size_t)(m_buf + (size_t)s * 256 * 256);
#pragma unroll 4
    for (int it = 0; it < 32; ++it) {
      int gch = tid + 256 * it;           // 16B chunk id (32 chunks / row)
      int r = gch >> 5, c = gch & 31;
      async_copy16(lds_byte_addr(smem + r * 528 + c * 16), gbase,
                   (unsigned)(gch * 16));
    }
    wait_async0();
  }
  __syncthreads();

  // ---- phase 1: Q,K,V,G projections (m @ W_h), WMMA f16 ------------------
  const float scale = 0.17677669529663687f;  // 1/sqrt(32)
  for (int mat = 0; mat < 4; ++mat) {
    const _Float16* wp =
        ((mat == 0) ? wq_t : (mat == 1) ? wk_t : (mat == 2) ? wv_t : wg_t) +
        h * 8192;                              // [d][c] for this head
    for (int i = 0; i < 4; ++i) {
      int t = w + 8 * i;                       // 0..31 tiles per matrix
      int mt = t >> 1, nt = t & 1;
      union { v8f v; float f[8]; } acc;
#pragma unroll
      for (int r = 0; r < 8; ++r) acc.f[r] = 0.f;
      for (int kk = 0; kk < 8; ++kk) {
        v16h a = load_frag_a(mp + mt * 16 * 264 + kk * 32, 264);
        v16h b = load_frag_b(wp + nt * 16 * 256 + kk * 32, 256);
        acc.v = wmma_f16(a, b, acc.v);
      }
      int row0 = mt * 16, n0 = nt * 16;
      if (mat == 0) {
#pragma unroll
        for (int r = 0; r < 8; ++r)
          qs[(row0 + r + 8 * hf) * 40 + n0 + lm] = (_Float16)(acc.f[r] * scale);
      } else if (mat == 1) {
#pragma unroll
        for (int r = 0; r < 8; ++r)
          ks[(row0 + r + 8 * hf) * 40 + n0 + lm] = (_Float16)acc.f[r];
      } else if (mat == 2) {
#pragma unroll
        for (int r = 0; r < 8; ++r)
          vt[(n0 + lm) * 264 + row0 + r + 8 * hf] = (_Float16)acc.f[r];
      } else {
        float bgv = bg[h * 32 + n0 + lm];
#pragma unroll
        for (int r = 0; r < 8; ++r) {
          float x = acc.f[r] + bgv;
          gate[(row0 + r + 8 * hf) * 33 + n0 + lm] = 1.f / (1.f + __expf(-x));
        }
      }
    }
  }
  __syncthreads();

  // ---- pass A: logits = Q K^T + z + mask_bias; store f16 into mp; rowmax -
  const float* zb = z_buf + h * 65536;
  const float* maskp = msa_mask + s * 256;
  for (int mi = 0; mi < 2; ++mi) {
    int mt = w * 2 + mi;
    int row0 = mt * 16;
    v16h qf = load_frag_a(qs + row0 * 40, 40);
    float rmax[8];
#pragma unroll
    for (int r = 0; r < 8; ++r) rmax[r] = -3.0e38f;
    for (int nt = 0; nt < 16; ++nt) {
      int n0 = nt * 16;
      v16h kf = load_frag_b(ks + n0 * 40, 40);
      union { v8f v; float f[8]; } acc;
#pragma unroll
      for (int r = 0; r < 8; ++r) acc.f[r] = 0.f;
      acc.v = wmma_f16(qf, kf, acc.v);
      int ng = n0 + lm;
      float mb = 1.0e9f * (maskp[ng] - 1.0f);
#pragma unroll
      for (int r = 0; r < 8; ++r) {
        int row = row0 + r + 8 * hf;
        float l = acc.f[r] + zb[row * 256 + ng] + mb;
        rmax[r] = fmaxf(rmax[r], l);
        mp[row * 264 + ng] = (_Float16)l;
      }
    }
#pragma unroll
    for (int r = 0; r < 8; ++r) {
      for (int off = 1; off < 16; off <<= 1)
        rmax[r] = fmaxf(rmax[r], __shfl_xor(rmax[r], off, 32));
    }
    if (lm == 0) {
#pragma unroll
      for (int r = 0; r < 8; ++r) rowmax[row0 + r + 8 * hf] = rmax[r];
    }
  }
  __syncthreads();

  // ---- pass B: softmax numerator + row sums (thread-per-row) -------------
  {
    int r = tid;
    float mx = rowmax[r];
    float sum = 0.f;
    _Float16* rowp = mp + r * 264;
    for (int i = 0; i < 256; ++i) {
      float p = __expf((float)rowp[i] - mx);
      sum += p;
      rowp[i] = (_Float16)p;
    }
    rowsum[r] = sum;
  }
  __syncthreads();

  // ---- pass C: O = softmax @ V, normalize, gate, store gated o -----------
  for (int i = 0; i < 4; ++i) {
    int t = w + 8 * i;                      // 0..31
    int mt = t >> 1, nt = t & 1;
    int row0 = mt * 16, n0 = nt * 16;
    union { v8f v; float f[8]; } acc;
#pragma unroll
    for (int r = 0; r < 8; ++r) acc.f[r] = 0.f;
    for (int kk = 0; kk < 8; ++kk) {
      v16h a = load_frag_a(mp + row0 * 264 + kk * 32, 264);
      v16h b = load_frag_b(vt + n0 * 264 + kk * 32, 264);
      acc.v = wmma_f16(a, b, acc.v);
    }
#pragma unroll
    for (int r = 0; r < 8; ++r) {
      int row = row0 + r + 8 * hf;
      int d = n0 + lm;
      float o = acc.f[r] / rowsum[row] * gate[row * 33 + d];
      o_buf[((size_t)(s * 256 + row) * 8 + h) * 32 + d] = (_Float16)o;
    }
  }
}

// ---------------------------------------------------------------------------
// Kernel 5: output projection.  out[sq][c] = o[sq][:] @ wo_t + bo  (f32 out)
// grid = 256 WGs, each: 128 rows x 256 cols, K=256.
// LDS: wo_t staged 256x264 f16 (135168) + A block 128x264 f16 (67584)
// ---------------------------------------------------------------------------
#define PROJ_LDS_BYTES 202752

__global__ __launch_bounds__(256) void outproj_kernel(
    const _Float16* __restrict__ o_buf, const _Float16* __restrict__ wo_t,
    const float* __restrict__ bo, float* __restrict__ out) {
  extern __shared__ uint4 dyn_smem[];
  char* smem = (char*)dyn_smem;
  _Float16* wl = (_Float16*)(smem);            // 256 x 264
  _Float16* al = (_Float16*)(smem + 135168);   // 128 x 264
  const int tid = threadIdx.x;
  const int w = tid >> 5, lane = tid & 31;
  const int lm = lane & 15, hf = lane >> 4;
  const int rowbase = blockIdx.x * 128;

  // async-stage wo_t (256x256) and the 128-row A block, both padded ld=264
  {
    const unsigned long long wbase = (unsigned long long)(size_t)wo_t;
#pragma unroll 4
    for (int it = 0; it < 32; ++it) {
      int gch = tid + 256 * it;
      int r = gch >> 5, c = gch & 31;
      async_copy16(lds_byte_addr(smem + r * 528 + c * 16), wbase,
                   (unsigned)(gch * 16));
    }
    const unsigned long long abase =
        (unsigned long long)(size_t)(o_buf + (size_t)rowbase * 256);
#pragma unroll 4
    for (int it = 0; it < 16; ++it) {
      int gch = tid + 256 * it;
      int r = gch >> 5, c = gch & 31;
      async_copy16(lds_byte_addr(smem + 135168 + r * 528 + c * 16), abase,
                   (unsigned)(gch * 16));
    }
    wait_async0();
  }
  __syncthreads();

  for (int i = 0; i < 16; ++i) {
    int t = w + 8 * i;                       // 0..127
    int mt = t >> 4, nt = t & 15;
    union { v8f v; float f[8]; } acc;
#pragma unroll
    for (int r = 0; r < 8; ++r) acc.f[r] = 0.f;
    for (int kk = 0; kk < 8; ++kk) {
      v16h a = load_frag_a(al + mt * 16 * 264 + kk * 32, 264);
      v16h b = load_frag_b(wl + nt * 16 * 264 + kk * 32, 264);
      acc.v = wmma_f16(a, b, acc.v);
    }
    float bv = bo[nt * 16 + lm];
#pragma unroll
    for (int r = 0; r < 8; ++r) {
      int row = rowbase + mt * 16 + r + 8 * hf;
      out[(size_t)row * 256 + nt * 16 + lm] = acc.f[r] + bv;
    }
  }
}

// ---------------------------------------------------------------------------
// Host launch
// ---------------------------------------------------------------------------
extern "C" void kernel_launch(void* const* d_in, const int* in_sizes, int n_in,
                              void* d_out, int out_size, void* d_ws,
                              size_t ws_size, hipStream_t stream) {
  const float* msa_act  = (const float*)d_in[0];
  const float* pair_act = (const float*)d_in[1];
  const float* msa_mask = (const float*)d_in[2];
  const float* qn_g     = (const float*)d_in[3];
  const float* qn_b     = (const float*)d_in[4];
  const float* pn_g     = (const float*)d_in[5];
  const float* pn_b     = (const float*)d_in[6];
  const float* w2d      = (const float*)d_in[7];
  const float* wq       = (const float*)d_in[8];
  const float* wk       = (const float*)d_in[9];
  const float* wv       = (const float*)d_in[10];
  const float* wg       = (const float*)d_in[11];
  const float* bg       = (const float*)d_in[12];
  const float* wo       = (const float*)d_in[13];
  const float* bo       = (const float*)d_in[14];

  char* ws = (char*)d_ws;
  _Float16* m_buf = (_Float16*)(ws);                  // 16,777,216 B
  float*    z_buf = (float*)(ws + 16777216);          //  2,097,152 B
  _Float16* wq_t  = (_Float16*)(ws + 18874368);       //    131,072 B
  _Float16* wk_t  = (_Float16*)(ws + 19005440);
  _Float16* wv_t  = (_Float16*)(ws + 19136512);
  _Float16* wg_t  = (_Float16*)(ws + 19267584);
  _Float16* wo_t  = (_Float16*)(ws + 19398656);
  _Float16* o_buf = (_Float16*)(ws + 19529728);       // 16,777,216 B

  prep_weights_kernel<<<1280, 256, 0, stream>>>(wq, wk, wv, wg, wo, wq_t, wk_t,
                                                wv_t, wg_t, wo_t);
  pair_bias_kernel<<<8192, 256, 0, stream>>>(pair_act, pn_g, pn_b, w2d, z_buf);
  ln_msa_kernel<<<4096, 256, 0, stream>>>(msa_act, qn_g, qn_b, m_buf);

  (void)hipFuncSetAttribute((const void*)attn_kernel,
                            hipFuncAttributeMaxDynamicSharedMemorySize,
                            ATTN_LDS_BYTES);
  attn_kernel<<<1024, 256, ATTN_LDS_BYTES, stream>>>(
      msa_mask, bg, m_buf, z_buf, wq_t, wk_t, wv_t, wg_t, o_buf);

  (void)hipFuncSetAttribute((const void*)outproj_kernel,
                            hipFuncAttributeMaxDynamicSharedMemorySize,
                            PROJ_LDS_BYTES);
  outproj_kernel<<<256, 256, PROJ_LDS_BYTES, stream>>>(o_buf, wo_t, bo,
                                                       (float*)d_out);
}
